// build_model_89807766159723
// MI455X (gfx1250) — compile-verified, matrix operand order checked by reading, not run on hardware
//
#include <hip/hip_runtime.h>
#include <hip/hip_bf16.h>
#include <math.h>

// ---------------------------------------------------------------------------
// Model constants (match reference)
// ---------------------------------------------------------------------------
#define BB     16
#define HIST   64
#define FUT    16
#define STATE  64
#define ACTION 32
#define DD     1024
#define HEADS  16
#define HDIM   64
#define LAYERS 8
#define MAXL   80

typedef __attribute__((ext_vector_type(16))) __bf16        v16bf;
typedef __attribute__((ext_vector_type(2)))  __bf16        v2bf;
typedef __attribute__((ext_vector_type(8)))  float         v8f;
typedef __attribute__((ext_vector_type(8)))  unsigned int  v8u;

union FragU {
  uint4 q[2];   // two 16-byte halves (maps to 2x ds_load_b128)
  v8u   u;
  v16bf b;
};

// pack two fp32 into a dword of two bf16; build_vector(fptrunc,fptrunc)
// pattern-matches to a single v_cvt_pk_bf16_f32 used directly.
__device__ __forceinline__ unsigned int pk2bf(float x, float y) {
  v2bf t;
  t.x = (__bf16)x;
  t.y = (__bf16)y;
  return __builtin_bit_cast(unsigned int, t);
}

// map a k-pair dword index (even kk) to (half, v) of the 16x32 bf16 fragment:
//   kk in {0,2,..,30}; VGPR v covers K { (v<4?0:16) + half*8 + (v&3)*2 }
__device__ __forceinline__ void kk_to_hv(int kk, int& half, int& v) {
  half = (kk >> 3) & 1;
  v = ((kk & 16) >> 2) + ((kk & 7) >> 1);
}

// ---------------------------------------------------------------------------
// Generic GEMM:  C[M x N] (fp32) = A[M x K] (fp32) * W (bf16, pre-swizzled)
// + bias, optionally C += (residual accumulate).
//
// W layout (built once by cvt_*_swz): [nt][kt][e], e in [0,1024) dwords where
//   e = ((cb*32 + half*16 + col16) << 3) + v   -- exactly the order the WMMA
// B-fragment is read from LDS, so GEMM B staging is a raw b128 copy.
//
// Preconditions: N multiple of 64, K multiple of 32, A rows 16B aligned.
// Only M is ragged; loads clamp the row index (stores are guarded).
//
// Tile: 128 rows x 64 cols, 256 threads = 8 waves; wave w owns rows
// [w*16,w*16+16) x 64 cols via 4 WMMA accumulators. Double-buffered LDS,
// one barrier per k-step.
// ---------------------------------------------------------------------------
__global__ __launch_bounds__(256) void gemm_bf16_wmma(
    const float* __restrict__ A, const unsigned int* __restrict__ Wt,
    const float* __restrict__ bias, float* __restrict__ C,
    int M, int N, int K, int accumulate)
{
  // A: 8 wave-fragments x 32 lanes x 8 dwords; B: 4 col-fragments x 32 x 8
  __shared__ unsigned int a_lds[2][8 * 32 * 8];
  __shared__ unsigned int b_lds[2][1024];

  const int tid  = threadIdx.x;
  const int wid  = tid >> 5;
  const int lane = tid & 31;
  const int half = lane >> 4;
  const int l16  = lane & 15;
  const int m0 = blockIdx.x * 128;
  const int ktiles = K >> 5;

  v8f zero = {0.f, 0.f, 0.f, 0.f, 0.f, 0.f, 0.f, 0.f};
  v8f acc[4];
  acc[0] = zero; acc[1] = zero; acc[2] = zero; acc[3] = zero;

  float4 aR[4];
  uint4 bR;

  // ---- staging helpers (branch-free; row index clamped for ragged M) ----
  auto loadA = [&](int k0) {
#pragma unroll
    for (int i = 0; i < 4; ++i) {
      int f = tid + (i << 8);            // 1024 float4 chunks: 128 rows x 8
      int row = f >> 3, k4 = f & 7;
      int gr = m0 + row; if (gr > M - 1) gr = M - 1;
      aR[i] = *(const float4*)(A + (size_t)gr * K + k0 + (k4 << 2));
    }
  };
  auto storeA = [&](int buf) {
    unsigned int* dst = &a_lds[buf][0];
#pragma unroll
    for (int i = 0; i < 4; ++i) {
      int f = tid + (i << 8);
      int row = f >> 3, k4 = f & 7;
      int wr = row >> 4, r16 = row & 15;
      int kk0 = k4 << 2, kk1 = kk0 + 2;
      int h0, v0, h1, v1;
      kk_to_hv(kk0, h0, v0);
      kk_to_hv(kk1, h1, v1);
      // kk0,kk1 share half and have consecutive v -> compiler merges to b64
      dst[((wr * 32 + h0 * 16 + r16) << 3) + v0] = pk2bf(aR[i].x, aR[i].y);
      dst[((wr * 32 + h1 * 16 + r16) << 3) + v1] = pk2bf(aR[i].z, aR[i].w);
    }
  };
  // B: pre-swizzled -> raw streaming copy, one b128 per thread per k-step
  auto loadB = [&](int kt) {
    bR = *(const uint4*)(Wt + ((((size_t)blockIdx.y * ktiles) + kt) << 10) +
                         (tid << 2));
  };
  auto storeB = [&](int buf) {
    *(uint4*)(&b_lds[buf][tid << 2]) = bR;
  };

  loadA(0); loadB(0);
  storeA(0); storeB(0);
  __syncthreads();

  for (int kt = 0; kt < ktiles; ++kt) {
    const int cur = kt & 1;
    if (kt + 1 < ktiles) { loadA((kt + 1) << 5); loadB(kt + 1); }

    const unsigned int* a32 = &a_lds[cur][0];
    const unsigned int* b32 = &b_lds[cur][0];

    // A fragment: 32 contiguous bytes per lane -> 2x ds_load_b128
    FragU af;
    {
      const uint4* ap = (const uint4*)(a32 + ((wid * 32 + lane) << 3));
      af.q[0] = ap[0];
      af.q[1] = ap[1];
    }
    // B fragments for the 4 column blocks
    FragU bfr[4];
#pragma unroll
    for (int cb = 0; cb < 4; ++cb) {
      const uint4* bp = (const uint4*)(b32 + ((cb * 32 + lane) << 3));
      bfr[cb].q[0] = bp[0];
      bfr[cb].q[1] = bp[1];
    }
#pragma unroll
    for (int cb = 0; cb < 4; ++cb)
      acc[cb] = __builtin_amdgcn_wmma_f32_16x16x32_bf16(
          false, af.b, false, bfr[cb].b, (short)0, acc[cb], false, false);

    if (kt + 1 < ktiles) { storeA(cur ^ 1); storeB(cur ^ 1); }
    __syncthreads();
  }

  // epilogue: C/D layout -> row = 8*half + g, col = l16 (per 16x16 block)
  const int n0 = blockIdx.y * 64;
#pragma unroll
  for (int cb = 0; cb < 4; ++cb) {
    int col = n0 + cb * 16 + l16;
    float bv = bias ? bias[col] : 0.f;
#pragma unroll
    for (int g = 0; g < 8; ++g) {
      int row = m0 + wid * 16 + half * 8 + g;
      if (row < M) {
        float val = acc[cb][g] + bv;
        size_t o = (size_t)row * N + col;
        if (accumulate) C[o] += val; else C[o] = val;
      }
    }
  }
}

// ---------------------------------------------------------------------------
// Weight conversion into the swizzled fragment-ready layout.
// dst dword idx decomposes as: l (layer), nt, kt, e; e encodes (cb,half,col,v)
// and maps to two consecutive K rows of the source at one column.
// ---------------------------------------------------------------------------
__device__ __forceinline__ void swz_decode(int r, int K, int& kt, int& e,
                                           int& nt) {
  int ktiles = K >> 5;
  nt = r / (ktiles << 10);
  int r2 = r - nt * (ktiles << 10);
  kt = r2 >> 10;
  e = r2 & 1023;
}

__device__ __forceinline__ void e_to_kn(int e, int& kk, int& nrel) {
  int v = e & 7, lanei = (e >> 3) & 31, cb = e >> 8;
  int h = lanei >> 4, cl = lanei & 15;
  kk = ((v & 4) << 2) + h * 8 + ((v & 3) << 1);
  nrel = cb * 16 + cl;
}

__global__ void cvt_swz(const float* __restrict__ src, unsigned int* __restrict__ dst,
                        int K, int N, int total) {
  int idx = blockIdx.x * 256 + threadIdx.x;
  if (idx >= total) return;
  int perL = (K * N) >> 1;
  int l = idx / perL;
  int r = idx - l * perL;
  int kt, e, nt;
  swz_decode(r, K, kt, e, nt);
  int kk, nrel;
  e_to_kn(e, kk, nrel);
  int k = (kt << 5) + kk;
  int n = (nt << 6) + nrel;
  const float* S = src + (size_t)l * K * N;
  dst[idx] = pk2bf(S[(size_t)k * N + n], S[(size_t)(k + 1) * N + n]);
}

__global__ void cvt_qkv_swz(const float* __restrict__ q, const float* __restrict__ kw,
                            const float* __restrict__ vw,
                            unsigned int* __restrict__ dst, int total) {
  // K = 1024, N = 3072 (q|k|v concatenated along N)
  int idx = blockIdx.x * 256 + threadIdx.x;
  if (idx >= total) return;
  const int perL = (1024 * 3072) >> 1;
  int l = idx / perL;
  int r = idx - l * perL;
  int kt, e, nt;
  swz_decode(r, 1024, kt, e, nt);
  int kk, nrel;
  e_to_kn(e, kk, nrel);
  int k = (kt << 5) + kk;
  int n = (nt << 6) + nrel;
  int sel = n >> 10, nn = n & 1023;
  const float* S = ((sel == 0) ? q : ((sel == 1) ? kw : vw)) +
                   (size_t)l * 1024 * 1024;
  dst[idx] = pk2bf(S[(size_t)k * 1024 + nn], S[(size_t)(k + 1) * 1024 + nn]);
}

__global__ void cvt_qkv_b(const float* __restrict__ q, const float* __restrict__ k,
                          const float* __restrict__ v, float* __restrict__ dst) {
  int idx = blockIdx.x * 256 + threadIdx.x;
  if (idx >= LAYERS * 3072) return;
  int n3 = idx % 3072, l = idx / 3072;
  int sel = n3 >> 10, nn = n3 & 1023;
  const float* s = (sel == 0) ? q : ((sel == 1) ? k : v);
  dst[idx] = s[l * 1024 + nn];
}

// ---------------------------------------------------------------------------
// Sinusoidal positional encoding (80 x 1024)
// ---------------------------------------------------------------------------
__global__ void pe_init(float* __restrict__ pe) {
  int idx = blockIdx.x * 256 + threadIdx.x;
  if (idx >= MAXL * DD) return;
  int d = idx & 1023, l = idx >> 10;
  float e2 = (float)(2 * (d >> 1)) / (float)DD;
  float denom = powf(10000.0f, e2);
  float a = (float)l / denom;
  pe[idx] = (d & 1) ? cosf(a) : sinf(a);
}

// ---------------------------------------------------------------------------
// LayerNorm over D=1024, one block (256 threads) per row
// ---------------------------------------------------------------------------
__device__ __forceinline__ void ln_row_dev(const float* __restrict__ x,
                                           float* __restrict__ y,
                                           const float* __restrict__ g,
                                           const float* __restrict__ b,
                                           float* red) {
  int t = threadIdx.x;
  float v0 = x[t], v1 = x[t + 256], v2 = x[t + 512], v3 = x[t + 768];
  red[t] = v0 + v1 + v2 + v3;
  __syncthreads();
  for (int off = 128; off > 0; off >>= 1) {
    if (t < off) red[t] += red[t + off];
    __syncthreads();
  }
  float mean = red[0] * (1.0f / 1024.0f);
  __syncthreads();
  float d0 = v0 - mean, d1 = v1 - mean, d2 = v2 - mean, d3 = v3 - mean;
  red[t] = d0 * d0 + d1 * d1 + d2 * d2 + d3 * d3;
  __syncthreads();
  for (int off = 128; off > 0; off >>= 1) {
    if (t < off) red[t] += red[t + off];
    __syncthreads();
  }
  float rstd = rsqrtf(red[0] * (1.0f / 1024.0f) + 1e-5f);
  y[t]       = d0 * rstd * g[t]       + b[t];
  y[t + 256] = d1 * rstd * g[t + 256] + b[t + 256];
  y[t + 512] = d2 * rstd * g[t + 512] + b[t + 512];
  y[t + 768] = d3 * rstd * g[t + 768] + b[t + 768];
}

__global__ __launch_bounds__(256) void ln_rows(const float* __restrict__ x,
                                               float* __restrict__ y,
                                               const float* __restrict__ g,
                                               const float* __restrict__ b) {
  __shared__ float red[256];
  size_t off = (size_t)blockIdx.x * DD;
  ln_row_dev(x + off, y + off, g, b, red);
}

// LN of the last token of each batch row (input stride L*D)
__global__ __launch_bounds__(256) void ln_last(const float* __restrict__ h,
                                               float* __restrict__ last,
                                               const float* __restrict__ g,
                                               const float* __restrict__ b, int L) {
  __shared__ float red[256];
  size_t in = ((size_t)blockIdx.x * L + (L - 1)) * DD;
  ln_row_dev(h + in, last + (size_t)blockIdx.x * DD, g, b, red);
}

// ---------------------------------------------------------------------------
// Elementwise kernels
// ---------------------------------------------------------------------------
__global__ void window_init(const float* __restrict__ hs, const float* __restrict__ ha,
                            float* __restrict__ window) {
  int idx = blockIdx.x * 256 + threadIdx.x;            // B*HIST*D
  if (idx >= BB * HIST * DD) return;
  int d = idx & 1023;
  int l = (idx >> 10) & 63;
  int b = idx >> 16;
  window[((size_t)(b * MAXL + l)) * DD + d] = hs[idx] + ha[idx];
}

__global__ void window_append(float* __restrict__ window,
                              const float* __restrict__ present,
                              const float* __restrict__ fa, int step) {
  int idx = blockIdx.x * 256 + threadIdx.x;            // B*D
  if (idx >= BB * DD) return;
  int d = idx & 1023, b = idx >> 10;
  window[((size_t)(b * MAXL + HIST + step)) * DD + d] =
      present[idx] + fa[((size_t)(b * FUT + step)) * DD + d];
}

__global__ void gelu_pe(const float* __restrict__ window, const float* __restrict__ pe,
                        float* __restrict__ h, int L) {
  int idx = blockIdx.x * 256 + threadIdx.x;            // B*L*D (compact)
  if (idx >= BB * L * DD) return;
  int d = idx & 1023;
  int l = (idx >> 10) % L;
  int b = (idx >> 10) / L;
  float x = window[((size_t)(b * MAXL + l)) * DD + d];
  float ge = 0.5f * x * (1.0f + erff(x * 0.70710678118654752f));
  h[idx] = ge + pe[l * DD + d];
}

// ---------------------------------------------------------------------------
// Causal attention: one block (128 threads) per (b, h, q)
// qkv layout: [b*L + l][3072] with q at 0, k at 1024, v at 2048
// ---------------------------------------------------------------------------
__global__ __launch_bounds__(128) void attention(const float* __restrict__ qkv,
                                                 float* __restrict__ attn, int L) {
  __shared__ float qrow[HDIM];
  __shared__ float sc[128];
  __shared__ float red[128];
  int bh = blockIdx.x;
  int b = bh >> 4, hh = bh & 15;
  int qi = blockIdx.y;
  int t = threadIdx.x;
  const float* base = qkv + (size_t)(b * L) * 3072;
  if (t < HDIM) qrow[t] = base[(size_t)qi * 3072 + hh * HDIM + t];
  __syncthreads();

  float s = -3.0e38f;
  if (t < L && t <= qi) {
    const float* krow = base + (size_t)t * 3072 + 1024 + hh * HDIM;
    float a = 0.f;
#pragma unroll
    for (int d = 0; d < HDIM; ++d) a += qrow[d] * krow[d];
    s = a * 0.125f;                  // 1/sqrt(64)
  }
  red[t] = s;
  __syncthreads();
  for (int off = 64; off > 0; off >>= 1) {
    if (t < off) red[t] = fmaxf(red[t], red[t + off]);
    __syncthreads();
  }
  float mx = red[0];
  __syncthreads();
  float e = (t < L && t <= qi) ? expf(s - mx) : 0.f;
  sc[t] = e;
  red[t] = e;
  __syncthreads();
  for (int off = 64; off > 0; off >>= 1) {
    if (t < off) red[t] += red[t + off];
    __syncthreads();
  }
  float inv = 1.0f / red[0];
  __syncthreads();
  if (t < HDIM) {
    float o = 0.f;
    for (int j = 0; j <= qi; ++j)
      o += sc[j] * base[(size_t)j * 3072 + 2048 + hh * HDIM + t];
    attn[((size_t)(b * L + qi)) * DD + hh * HDIM + t] = o * inv;
  }
}

// ---------------------------------------------------------------------------
// Heads: reward = tanh(last @ W_r + b_r); state = scaled_mish(last @ W_s + b_s)
// ---------------------------------------------------------------------------
__global__ __launch_bounds__(256) void reward_head(const float* __restrict__ last,
                                                   const float* __restrict__ Wr,
                                                   const float* __restrict__ br,
                                                   float* __restrict__ out_r, int step) {
  __shared__ float red[256];
  int b = blockIdx.x, t = threadIdx.x;
  const float* x = last + (size_t)b * DD;
  float s = x[t] * Wr[t] + x[t + 256] * Wr[t + 256] +
            x[t + 512] * Wr[t + 512] + x[t + 768] * Wr[t + 768];
  red[t] = s;
  __syncthreads();
  for (int off = 128; off > 0; off >>= 1) {
    if (t < off) red[t] += red[t + off];
    __syncthreads();
  }
  if (t == 0) out_r[b * FUT + step] = tanhf(red[0] + br[0]);
}

__global__ __launch_bounds__(64) void state_head(const float* __restrict__ last,
                                                 const float* __restrict__ Ws,
                                                 const float* __restrict__ bs,
                                                 float* __restrict__ out_s,
                                                 float* __restrict__ scur, int step) {
  int b = blockIdx.x, n = threadIdx.x;
  const float* x = last + (size_t)b * DD;
  float a = 0.f;
  for (int k = 0; k < DD; ++k) a += x[k] * Ws[k * STATE + n];
  float xv = a + bs[n];
  float sp = (xv > 20.f) ? xv : log1pf(expf(xv));
  float y = 0.95f * tanhf(xv * tanhf(sp));
  out_s[((size_t)(b * FUT + step)) * STATE + n] = y;
  scur[b * STATE + n] = y;
}

// ---------------------------------------------------------------------------
// Host orchestration
// ---------------------------------------------------------------------------
extern "C" void kernel_launch(void* const* d_in, const int* in_sizes, int n_in,
                              void* d_out, int out_size, void* d_ws, size_t ws_size,
                              hipStream_t stream) {
  const float* history_s = (const float*)d_in[0];
  const float* history_a = (const float*)d_in[1];
  const float* present_s = (const float*)d_in[2];
  /* future_s (d_in[3]) unused by the reference */
  const float* future_a  = (const float*)d_in[4];
  const float* W_state = (const float*)d_in[5];
  const float* b_state = (const float*)d_in[6];
  const float* W_action = (const float*)d_in[7];
  const float* b_action = (const float*)d_in[8];
  const float* sn_g = (const float*)d_in[9];
  const float* sn_b = (const float*)d_in[10];
  const float* an_g = (const float*)d_in[11];
  const float* an_b = (const float*)d_in[12];
  const float* ln1_g = (const float*)d_in[13];
  const float* ln1_b = (const float*)d_in[14];
  const float* Wq = (const float*)d_in[15];
  const float* bq = (const float*)d_in[16];
  const float* Wk = (const float*)d_in[17];
  const float* bk = (const float*)d_in[18];
  const float* Wv = (const float*)d_in[19];
  const float* bv = (const float*)d_in[20];
  const float* Wo = (const float*)d_in[21];
  const float* bo = (const float*)d_in[22];
  const float* ln2_g = (const float*)d_in[23];
  const float* ln2_b = (const float*)d_in[24];
  const float* Wfc = (const float*)d_in[25];
  const float* bfc = (const float*)d_in[26];
  const float* tn_g = (const float*)d_in[27];
  const float* tn_b = (const float*)d_in[28];
  const float* W_r = (const float*)d_in[29];
  const float* b_r = (const float*)d_in[30];
  const float* W_s = (const float*)d_in[31];
  const float* b_s = (const float*)d_in[32];

  float* out_r = (float*)d_out;                 // (B, FUT, 1) flat
  float* out_s = (float*)d_out + BB * FUT;      // (B, FUT, 64) flat

  // ---- workspace bump allocator ----
  char* p = (char*)d_ws;
  auto alloc = [&](size_t bytes) -> void* {
    void* r = (void*)p;
    p += (bytes + 255) & ~((size_t)255);
    return r;
  };
  // swizzled bf16 weights (dword units: K*N/2 per layer)
  unsigned int* Wqkv_t = (unsigned int*)alloc((size_t)LAYERS * DD * 3072 * 2);
  unsigned int* Wo_t   = (unsigned int*)alloc((size_t)LAYERS * DD * DD * 2);
  unsigned int* Wfc_t  = (unsigned int*)alloc((size_t)LAYERS * DD * DD * 2);
  unsigned int* Wst_t  = (unsigned int*)alloc((size_t)STATE * DD * 2);
  unsigned int* Wac_t  = (unsigned int*)alloc((size_t)ACTION * DD * 2);
  float* bqkv   = (float*)alloc((size_t)LAYERS * 3072 * 4);
  float* pe     = (float*)alloc((size_t)MAXL * DD * 4);
  float* window = (float*)alloc((size_t)BB * MAXL * DD * 4);
  float* hbuf   = (float*)alloc((size_t)BB * MAXL * DD * 4);
  float* hn     = (float*)alloc((size_t)BB * MAXL * DD * 4);
  float* qkv    = (float*)alloc((size_t)BB * MAXL * 3072 * 4);
  float* attn   = (float*)alloc((size_t)BB * MAXL * DD * 4);
  float* fa     = (float*)alloc((size_t)BB * FUT * DD * 4);
  float* present = (float*)alloc((size_t)BB * DD * 4);
  float* tmp     = (float*)alloc((size_t)BB * HIST * DD * 4);
  float* last    = (float*)alloc((size_t)BB * DD * 4);
  float* scur    = (float*)alloc((size_t)BB * STATE * 4);

  auto cdiv = [](int a, int b) { return (a + b - 1) / b; };
  auto gemm = [&](const float* A, const unsigned int* Wt, const float* bias,
                  float* C, int M, int N, int K, int accum) {
    dim3 g(cdiv(M, 128), N / 64);
    gemm_bf16_wmma<<<g, 256, 0, stream>>>(A, Wt, bias, C, M, N, K, accum);
  };
  auto lnk = [&](const float* x, float* y, const float* g, const float* b, int rows) {
    ln_rows<<<rows, 256, 0, stream>>>(x, y, g, b);
  };

  // ---- one-time (per call) prep: swizzled bf16 weights, PE table ----
  {
    int n = LAYERS * DD * 3072 / 2;
    cvt_qkv_swz<<<cdiv(n, 256), 256, 0, stream>>>(Wq, Wk, Wv, Wqkv_t, n);
    cvt_qkv_b<<<cdiv(LAYERS * 3072, 256), 256, 0, stream>>>(bq, bk, bv, bqkv);
    n = LAYERS * DD * DD / 2;
    cvt_swz<<<cdiv(n, 256), 256, 0, stream>>>(Wo, Wo_t, DD, DD, n);
    cvt_swz<<<cdiv(n, 256), 256, 0, stream>>>(Wfc, Wfc_t, DD, DD, n);
    n = STATE * DD / 2;
    cvt_swz<<<cdiv(n, 256), 256, 0, stream>>>(W_state, Wst_t, STATE, DD, n);
    n = ACTION * DD / 2;
    cvt_swz<<<cdiv(n, 256), 256, 0, stream>>>(W_action, Wac_t, ACTION, DD, n);
    pe_init<<<cdiv(MAXL * DD, 256), 256, 0, stream>>>(pe);
  }

  // ---- embeddings ----
  gemm(history_s, Wst_t, b_state, hbuf, BB * HIST, DD, STATE, 0);
  lnk(hbuf, hbuf, sn_g, sn_b, BB * HIST);
  gemm(history_a, Wac_t, b_action, tmp, BB * HIST, DD, ACTION, 0);
  lnk(tmp, tmp, an_g, an_b, BB * HIST);
  window_init<<<cdiv(BB * HIST * DD, 256), 256, 0, stream>>>(hbuf, tmp, window);

  gemm(future_a, Wac_t, b_action, fa, BB * FUT, DD, ACTION, 0);
  lnk(fa, fa, an_g, an_b, BB * FUT);

  gemm(present_s, Wst_t, b_state, tmp, BB, DD, STATE, 0);
  lnk(tmp, present, sn_g, sn_b, BB);

  // ---- autoregressive rollout ----
  for (int i = 0; i < FUT; ++i) {
    int L = HIST + 1 + i;
    window_append<<<cdiv(BB * DD, 256), 256, 0, stream>>>(window, present, fa, i);
    gelu_pe<<<cdiv(BB * L * DD, 256), 256, 0, stream>>>(window, pe, hbuf, L);

    for (int j = 0; j < LAYERS; ++j) {
      lnk(hbuf, hn, ln1_g + j * DD, ln1_b + j * DD, BB * L);
      gemm(hn, Wqkv_t + (size_t)j * DD * 3072 / 2, bqkv + j * 3072, qkv,
           BB * L, 3072, DD, 0);
      attention<<<dim3(BB * HEADS, L), 128, 0, stream>>>(qkv, attn, L);
      gemm(attn, Wo_t + (size_t)j * DD * DD / 2, bo + j * DD, hbuf,
           BB * L, DD, DD, 1);                               // h += attn @ Wo + bo
      lnk(hbuf, hn, ln2_g + j * DD, ln2_b + j * DD, BB * L);
      gemm(hn, Wfc_t + (size_t)j * DD * DD / 2, bfc + j * DD, hbuf,
           BB * L, DD, DD, 1);                               // h += ln2(h) @ Wfc + bfc
    }

    ln_last<<<BB, 256, 0, stream>>>(hbuf, last, tn_g, tn_b, L);
    reward_head<<<BB, 256, 0, stream>>>(last, W_r, b_r, out_r, i);
    state_head<<<BB, 64, 0, stream>>>(last, W_s, b_s, out_s, scur, i);

    gemm(scur, Wst_t, b_state, tmp, BB, DD, STATE, 0);
    lnk(tmp, present, sn_g, sn_b, BB);
  }
}
